// ProcessNeurons_75007308857724
// MI455X (gfx1250) — matrix operand order repaired and checked by reading, not compiled, not run on hardware
//
#include <hip/hip_runtime.h>
#include <hip/hip_bf16.h>

// Problem dimensions (fixed by the reference).
#define B_    8
#define S_    2048
#define KIN_  512
#define D_    1024
#define NIN_  4096
#define P_    2048
#define KSEL_ 256

// Padded LDS row stride (elements): 32 data + 8 pad -> 80B = 20-bank stride,
// 20*i mod 64 is distinct for i=0..15 => conflict-free fragment row reads.
#define LDSL  40

typedef __attribute__((ext_vector_type(4)))  __bf16 v4bf;
typedef __attribute__((ext_vector_type(8)))  __bf16 v8bf;
typedef __attribute__((ext_vector_type(16))) __bf16 v16bf;
typedef __attribute__((ext_vector_type(8)))  float  v8f;
// Exact pointee type of the async-load builtin's pointer params (per hipcc
// diagnostic: "__attribute__((__vector_size__(4 * sizeof(int)))) int __device__ *").
typedef int v4i_ __attribute__((vector_size(16)));

__device__ __forceinline__ float gelu_erf(float x) {
    return 0.5f * x * (1.0f + erff(x * 0.70710678118654752f));
}

// ---------------------------------------------------------------------------
// CDNA5 async global->LDS copy (16B per lane), tracked by ASYNCcnt.
// ---------------------------------------------------------------------------
__device__ __forceinline__ void async_cp16(const __bf16* g, __bf16* l) {
#if __has_builtin(__builtin_amdgcn_global_load_async_to_lds_b128)
    __builtin_amdgcn_global_load_async_to_lds_b128(
        (__attribute__((address_space(1))) v4i_*)(__bf16*)g,
        (__attribute__((address_space(3))) v4i_*)l, 0, 0);
#else
    unsigned loff = (unsigned)(unsigned long long)
        (__attribute__((address_space(3))) void*)l;
    asm volatile("global_load_async_to_lds_b128 %0, %1, off"
                 :: "v"(loff), "v"(g) : "memory");
#endif
}

template <int N>
__device__ __forceinline__ void s_wait_async() {
#if __has_builtin(__builtin_amdgcn_s_wait_asynccnt)
    __builtin_amdgcn_s_wait_asynccnt((unsigned short)N);
#else
    asm volatile("s_wait_asynccnt %0" :: "i"(N));
#endif
}

// Issue one double-buffer stage: A tile 128x32, B tile 256x32 (bf16),
// 6 async B128 transfers per thread (2 for A, 4 for B).
__device__ __forceinline__ void stage_tiles(const __bf16* __restrict__ Ab,
                                            const __bf16* __restrict__ Bb,
                                            int ldg, int kk,
                                            __bf16* lA, __bf16* lB, int tid) {
#pragma unroll
    for (int it = 0; it < 2; ++it) {          // 128 rows * 4 chunks / 256 thr
        const int t = tid + 256 * it;
        const int row = t >> 2, ch = t & 3;
        async_cp16(Ab + (size_t)row * ldg + kk + ch * 8, lA + row * LDSL + ch * 8);
    }
#pragma unroll
    for (int it = 0; it < 4; ++it) {          // 256 rows * 4 chunks / 256 thr
        const int t = tid + 256 * it;
        const int row = t >> 2, ch = t & 3;
        async_cp16(Bb + (size_t)row * ldg + kk + ch * 8, lB + row * LDSL + ch * 8);
    }
}

// ---------------------------------------------------------------------------
// WMMA fragment loaders from an LDS tile (bf16 16x16x32, wave32),
// per ISA §7.12.2 layouts.
// A (16x32): lane L holds row m=(L&15); k = 8*(L>>4) + (e<8 ? e : e+8)
//            -> two 16B ds loads per lane.
// B (32x16) from [N][K] rows: lane L holds col n=(L&15), k = 8*(L>>4) + (e<8 ? e : e+8)
//            -> two 16B ds loads per lane.
// C/D: lane L, vgpr r -> m = 8*(L>>4)+r, n = (L&15).
// ---------------------------------------------------------------------------
__device__ __forceinline__ v16bf lds_a_frag(const __bf16* tile, int row0, int lane) {
    const __bf16* p = tile + (row0 + (lane & 15)) * LDSL + 8 * (lane >> 4);
    v8bf lo = *(const v8bf*)p;
    v8bf hi = *(const v8bf*)(p + 16);
    return __builtin_shufflevector(lo, hi, 0, 1, 2, 3, 4, 5, 6, 7,
                                   8, 9, 10, 11, 12, 13, 14, 15);
}

__device__ __forceinline__ v16bf lds_b_frag(const __bf16* tile, int col0, int lane) {
    const __bf16* p = tile + (col0 + (lane & 15)) * LDSL + 8 * (lane >> 4);
    v8bf lo = *(const v8bf*)p;
    v8bf hi = *(const v8bf*)(p + 16);
    return __builtin_shufflevector(lo, hi, 0, 1, 2, 3, 4, 5, 6, 7,
                                   8, 9, 10, 11, 12, 13, 14, 15);
}

// ---------------------------------------------------------------------------
// Stage 0: f32 -> bf16 convert of activations [B,S,KIN]
// ---------------------------------------------------------------------------
__global__ void k_convertA(const float* __restrict__ src, __bf16* __restrict__ dst) {
    size_t i = (size_t)blockIdx.x * blockDim.x + threadIdx.x; // quad id
    const float4 v = ((const float4*)src)[i];
    v4bf o;
    o[0] = (__bf16)v.x; o[1] = (__bf16)v.y; o[2] = (__bf16)v.z; o[3] = (__bf16)v.w;
    *(v4bf*)(dst + 4 * i) = o;
}

// Stage 0b: gather combination weights per batch + convert:
//   W[b][p][j] = bf16(cw[p][ idx[b][j] ])
__global__ void k_gatherW(const float* __restrict__ cw, const int* __restrict__ idx,
                          __bf16* __restrict__ dst) {
    size_t i = (size_t)blockIdx.x * blockDim.x + threadIdx.x; // over B*P*KIN
    int    j  = (int)(i % KIN_);
    size_t bp = i / KIN_;
    int    p  = (int)(bp % P_);
    int    b  = (int)(bp / P_);
    int    col = idx[(size_t)b * KIN_ + j];
    dst[i] = (__bf16)cw[(size_t)p * NIN_ + col];
}

__global__ void k_zero_scores(float* __restrict__ s) {
    s[blockIdx.x * 256 + threadIdx.x] = 0.0f;
}

// ---------------------------------------------------------------------------
// Stage 1: per-batch GEMM  act[b,s,p] = GELU( A[b,s,:] . W[b,p,:] )
// 8 waves/block; block tile 128(S) x 256(P) x 32(K); 64x64 per wave.
// Operands staged to LDS with double-buffered async B128 copies.
// Also accumulates column sums (gating scores) via f32 global atomics.
// ---------------------------------------------------------------------------
__global__ void __launch_bounds__(256, 2)
k_gemm1(const __bf16* __restrict__ A, const __bf16* __restrict__ W,
        __bf16* __restrict__ act, float* __restrict__ scores) {
    __shared__ __bf16 lA[2][128 * LDSL];
    __shared__ __bf16 lB[2][256 * LDSL];

    const int tid  = threadIdx.x;
    const int lane = tid & 31;
    const int wave = tid >> 5;
    const int b    = blockIdx.z;
    const int sblk = blockIdx.y * 128;
    const int pblk = blockIdx.x * 256;
    const int wrow = (wave >> 2) * 64;   // wave tile row inside block tile
    const int wcol = (wave & 3) * 64;    // wave tile col inside block tile

    const __bf16* Ab = A + ((size_t)b * S_ + sblk) * KIN_;
    const __bf16* Wb = W + ((size_t)b * P_ + pblk) * KIN_;

    v8f acc[4][4] = {};

    stage_tiles(Ab, Wb, KIN_, 0, lA[0], lB[0], tid);

    const int nsteps = KIN_ / 32; // 16
    for (int step = 0; step < nsteps; ++step) {
        const int cur = step & 1;
        if (step + 1 < nsteps) {
            stage_tiles(Ab, Wb, KIN_, (step + 1) * 32, lA[1 - cur], lB[1 - cur], tid);
            s_wait_async<6>();   // previous stage complete (in-order ASYNCcnt)
        } else {
            s_wait_async<0>();
        }
        __syncthreads();         // all waves' copies visible

        v16bf af[4], bfr[4];
#pragma unroll
        for (int t = 0; t < 4; ++t) af[t]  = lds_a_frag(lA[cur], wrow + 16 * t, lane);
#pragma unroll
        for (int t = 0; t < 4; ++t) bfr[t] = lds_b_frag(lB[cur], wcol + 16 * t, lane);
#pragma unroll
        for (int i = 0; i < 4; ++i)
#pragma unroll
            for (int j = 0; j < 4; ++j)
                acc[i][j] = __builtin_amdgcn_wmma_f32_16x16x32_bf16(
                    false, af[i], false, bfr[j], (short)0, acc[i][j], false, false);

        __syncthreads();         // reads done before buffer is refilled
    }

    __bf16* actb = act + (size_t)b * S_ * P_;
#pragma unroll
    for (int j = 0; j < 4; ++j) {
        const int n = pblk + wcol + 16 * j + (lane & 15);
        float colsum = 0.0f;
#pragma unroll
        for (int i = 0; i < 4; ++i) {
            const int m0 = sblk + wrow + 16 * i + 8 * (lane >> 4);
            v8f c = acc[i][j];
#pragma unroll
            for (int r = 0; r < 8; ++r) {
                float g = gelu_erf(c[r]);
                actb[(size_t)(m0 + r) * P_ + n] = (__bf16)g;
                colsum += g;
            }
        }
        atomicAdd(&scores[(size_t)b * P_ + n], colsum);
    }
}

// ---------------------------------------------------------------------------
// Stage 2: exact top-256 of 2048 scores per batch via rank counting.
// (Only the selected set matters: the final einsum sums over k.)
// ---------------------------------------------------------------------------
__global__ void k_topk(const float* __restrict__ scores, int* __restrict__ pidx) {
    const int b = blockIdx.x;
    __shared__ float s[P_];
    for (int i = threadIdx.x; i < P_; i += 256) s[i] = scores[(size_t)b * P_ + i];
    __syncthreads();
    for (int p = threadIdx.x; p < P_; p += 256) {
        const float v = s[p];
        int rank = 0;
        for (int q = 0; q < P_; ++q) {
            const float u = s[q];
            rank += (u > v) || (u == v && q < p);
        }
        if (rank < KSEL_) pidx[(size_t)b * KSEL_ + rank] = p;
    }
}

// Stage 3a: Ag[b][s][k] = act[b][s][pidx[b][k]]   (k-contiguous A operand)
__global__ void k_gatherAg(const __bf16* __restrict__ act, const int* __restrict__ pidx,
                           __bf16* __restrict__ dst) {
    size_t i = (size_t)blockIdx.x * blockDim.x + threadIdx.x; // over B*S*KSEL
    int    kk = (int)(i % KSEL_);
    size_t bs = i / KSEL_;
    int    s  = (int)(bs % S_);
    int    b  = (int)(bs / S_);
    int    p  = pidx[(size_t)b * KSEL_ + kk];
    dst[i] = act[((size_t)b * S_ + s) * P_ + p];
}

// Stage 3b: PjT[b][d][k] = bf16(proj[pidx[b][k]][d])  ([N][K] B operand)
__global__ void k_gatherPjT(const float* __restrict__ proj, const int* __restrict__ pidx,
                            __bf16* __restrict__ dst) {
    size_t i = (size_t)blockIdx.x * blockDim.x + threadIdx.x; // over B*D*KSEL
    int    kk = (int)(i % KSEL_);
    size_t bd = i / KSEL_;
    int    d  = (int)(bd % D_);
    int    b  = (int)(bd / D_);
    int    p  = pidx[(size_t)b * KSEL_ + kk];
    dst[i] = (__bf16)proj[(size_t)p * D_ + d];
}

// ---------------------------------------------------------------------------
// Stage 4: per-batch GEMM  out[b,s,d] = Ag[b,s,:] . PjT[b,d,:]  (f32 out)
// Same async-LDS double-buffered pipeline; K loop = 8 steps.
// ---------------------------------------------------------------------------
__global__ void __launch_bounds__(256, 2)
k_gemm2(const __bf16* __restrict__ Ag, const __bf16* __restrict__ PjT,
        float* __restrict__ out) {
    __shared__ __bf16 lA[2][128 * LDSL];
    __shared__ __bf16 lB[2][256 * LDSL];

    const int tid  = threadIdx.x;
    const int lane = tid & 31;
    const int wave = tid >> 5;
    const int b    = blockIdx.z;
    const int sblk = blockIdx.y * 128;
    const int dblk = blockIdx.x * 256;
    const int wrow = (wave >> 2) * 64;
    const int wcol = (wave & 3) * 64;

    const __bf16* Ab = Ag  + ((size_t)b * S_ + sblk) * KSEL_;
    const __bf16* Bb = PjT + ((size_t)b * D_ + dblk) * KSEL_;

    v8f acc[4][4] = {};

    stage_tiles(Ab, Bb, KSEL_, 0, lA[0], lB[0], tid);

    const int nsteps = KSEL_ / 32; // 8
    for (int step = 0; step < nsteps; ++step) {
        const int cur = step & 1;
        if (step + 1 < nsteps) {
            stage_tiles(Ab, Bb, KSEL_, (step + 1) * 32, lA[1 - cur], lB[1 - cur], tid);
            s_wait_async<6>();
        } else {
            s_wait_async<0>();
        }
        __syncthreads();

        v16bf af[4], bfr[4];
#pragma unroll
        for (int t = 0; t < 4; ++t) af[t]  = lds_a_frag(lA[cur], wrow + 16 * t, lane);
#pragma unroll
        for (int t = 0; t < 4; ++t) bfr[t] = lds_b_frag(lB[cur], wcol + 16 * t, lane);
#pragma unroll
        for (int i = 0; i < 4; ++i)
#pragma unroll
            for (int j = 0; j < 4; ++j)
                acc[i][j] = __builtin_amdgcn_wmma_f32_16x16x32_bf16(
                    false, af[i], false, bfr[j], (short)0, acc[i][j], false, false);

        __syncthreads();
    }

    float* ob = out + (size_t)b * S_ * D_;
#pragma unroll
    for (int j = 0; j < 4; ++j) {
        const int n = dblk + wcol + 16 * j + (lane & 15);
#pragma unroll
        for (int i = 0; i < 4; ++i) {
            const int m0 = sblk + wrow + 16 * i + 8 * (lane >> 4);
            v8f c = acc[i][j];
#pragma unroll
            for (int r = 0; r < 8; ++r)
                ob[(size_t)(m0 + r) * D_ + n] = c[r];
        }
    }
}

// ---------------------------------------------------------------------------
extern "C" void kernel_launch(void* const* d_in, const int* in_sizes, int n_in,
                              void* d_out, int out_size, void* d_ws, size_t ws_size,
                              hipStream_t stream) {
    (void)in_sizes; (void)n_in; (void)out_size; (void)ws_size;

    const float* sa   = (const float*)d_in[0]; // [B,S,KIN] f32
    const int*   sidx = (const int*)d_in[1];   // [B,KIN]
    const float* cw   = (const float*)d_in[3]; // [P,NIN]   f32
    const float* op   = (const float*)d_in[4]; // [P,D]     f32
    float*       out  = (float*)d_out;         // [B,S,D]   f32

    char*  ws  = (char*)d_ws;
    size_t off = 0;
    auto alloc = [&](size_t bytes) -> void* {
        void* p = ws + off;
        off = (off + bytes + 255) & ~(size_t)255;
        return p;
    };

    __bf16* Abf   = (__bf16*)alloc((size_t)B_ * S_ * KIN_  * 2); // 16 MB
    __bf16* Wbf   = (__bf16*)alloc((size_t)B_ * P_ * KIN_  * 2); // 16 MB
    __bf16* actbf = (__bf16*)alloc((size_t)B_ * S_ * P_    * 2); // 64 MB
    float*  sc    = (float*) alloc((size_t)B_ * P_         * 4); // 64 KB
    int*    pidx  = (int*)   alloc((size_t)B_ * KSEL_      * 4); //  8 KB
    __bf16* Agbf  = (__bf16*)alloc((size_t)B_ * S_ * KSEL_ * 2); //  8 MB
    __bf16* PjT   = (__bf16*)alloc((size_t)B_ * D_ * KSEL_ * 2); //  4 MB

    // Stage 0: bf16 staging (operands then live in the 192MB L2 for the GEMMs)
    k_convertA<<<dim3((B_ * S_ * KIN_ / 4) / 256), dim3(256), 0, stream>>>(sa, Abf);
    k_gatherW<<<dim3((B_ * (size_t)P_ * KIN_) / 256), dim3(256), 0, stream>>>(cw, sidx, Wbf);
    k_zero_scores<<<dim3(B_ * P_ / 256), dim3(256), 0, stream>>>(sc);

    // Stage 1: GEMM1 + GELU + score accumulation
    k_gemm1<<<dim3(P_ / 256, S_ / 128, B_), dim3(256), 0, stream>>>(Abf, Wbf, actbf, sc);

    // Stage 2: exact top-k set per batch
    k_topk<<<dim3(B_), dim3(256), 0, stream>>>(sc, pidx);

    // Stage 3: gather GEMM2 operands (k-contiguous)
    k_gatherAg<<<dim3((B_ * (size_t)S_ * KSEL_) / 256), dim3(256), 0, stream>>>(actbf, pidx, Agbf);
    k_gatherPjT<<<dim3((B_ * (size_t)D_ * KSEL_) / 256), dim3(256), 0, stream>>>(op, pidx, PjT);

    // Stage 4: GEMM2 -> f32 output
    k_gemm2<<<dim3(D_ / 256, S_ / 128, B_), dim3(256), 0, stream>>>(Agbf, PjT, out);
}
